// Item2Item_65867618452128
// MI455X (gfx1250) — matrix operand order repaired and checked by reading, not compiled, not run on hardware
//
#include <hip/hip_runtime.h>

typedef __attribute__((ext_vector_type(16))) __bf16 v16bf;
typedef __attribute__((ext_vector_type(8)))  float  v8f;

namespace {
constexpr int Bn = 512;
constexpr int Sn = 2048;
constexpr int En = 64;
constexpr int Pn = 32;
constexpr int Hn = 64;
constexpr int WAVES = 8;
constexpr int ROWS_PER_WAVE  = Sn / WAVES;         // 256
constexpr int TILES_PER_WAVE = ROWS_PER_WAVE / 16; // 16
constexpr float NEGV   = -4294967295.0f;           // -2^32 + 1 (reference pad)
constexpr float LOG2E  = 1.4426950408889634f;
}

// exp(x) via hardware v_exp_f32 (exp2). Branchless; x*log2e underflows to 0 for
// the huge-negative masked scores, which is exactly the softmax semantics we need.
__device__ __forceinline__ float fast_exp(float x) {
    return __builtin_amdgcn_exp2f(x * LOG2E);
}

// tanh: prefer the gfx1250 hardware V_TANH_F32 (TRANS op -> co-executes with WMMA).
// Fallback is branchless: tanh(x) = copysign(1 - 2/(exp2(2*log2e*|x|)+1), x);
// exp2 -> +inf for large |x| gives rcp -> 0 -> result +-1, no range checks needed.
__device__ __forceinline__ float fast_tanh(float x) {
#if __has_builtin(__builtin_amdgcn_tanhf)
    return __builtin_amdgcn_tanhf(x);
#else
    float ax = __builtin_fabsf(x);
    float e  = __builtin_amdgcn_exp2f(ax * (2.0f * LOG2E));
    float t  = __builtin_fmaf(-2.0f, __builtin_amdgcn_rcpf(e + 1.0f), 1.0f);
    return __builtin_copysignf(t, x);
#endif
}

__global__ __launch_bounds__(256)
void i2i_attention_kernel(const float* __restrict__ items,
                          const float* __restrict__ pos,
                          const float* __restrict__ tgt,
                          const float* __restrict__ Wp,
                          const float* __restrict__ We,
                          const float* __restrict__ Wc,
                          const float* __restrict__ bias,
                          const float* __restrict__ zvec,
                          const int*   __restrict__ seq_len,
                          float* __restrict__ out_u,
                          float* __restrict__ out_r)
{
    __shared__ float s_cb[Hn];
    __shared__ float s_m[WAVES], s_d[WAVES], s_r[WAVES];
    __shared__ float s_u[WAVES][En];

    const int b    = blockIdx.x;
    const int tid  = threadIdx.x;
    const int wave = tid >> 5;
    const int lane = tid & 31;
    const int l16  = lane & 15;
    const int hi   = lane >> 4;      // 0: lanes 0-15, 1: lanes 16-31

    // cb[h] = bias[h] + target_item[b,:] @ Wc[:,h]   (tiny, done once per block)
    if (tid < Hn) {
        float acc = bias[tid];
        const float* tb = tgt + (size_t)b * En;
        #pragma unroll 8
        for (int e = 0; e < En; ++e) acc += tb[e] * Wc[e * Hn + tid];
        s_cb[tid] = acc;
    }
    __syncthreads();

    const int slen = seq_len[b];

    float zv[4], cbv[4];
    #pragma unroll
    for (int nt = 0; nt < 4; ++nt) {
        zv[nt]  = zvec[nt * 16 + l16];
        cbv[nt] = s_cb[nt * 16 + l16];
    }

    // Stationary B fragments (bf16). WMMA B layout (KxN = 32x16 slab):
    //   column n=l16; elem j<8 : K = hi*8 + j ; elem j>=8 : K = 16 + hi*8 + (j-8)
    v16bf fWe[2][4];   // We 64x64 -> 2 K-slabs x 4 N-tiles
    v16bf fWp[4];      // Wp 32x64 -> 1 K-slab  x 4 N-tiles
    {
        const int kb = hi * 8;
        #pragma unroll
        for (int nt = 0; nt < 4; ++nt) {
            const int n = nt * 16 + l16;
            #pragma unroll
            for (int kt = 0; kt < 2; ++kt) {
                v16bf f;
                #pragma unroll
                for (int j = 0; j < 16; ++j)
                    f[j] = (__bf16)We[(kt * 32 + kb + j + (j >= 8 ? 8 : 0)) * Hn + n];
                fWe[kt][nt] = f;
            }
            v16bf g;
            #pragma unroll
            for (int j = 0; j < 16; ++j)
                g[j] = (__bf16)Wp[(kb + j + (j >= 8 ? 8 : 0)) * Hn + n];
            fWp[nt] = g;
        }
    }

    // online-softmax running state (single pass over the sequence)
    float m = NEGV, d = 0.0f, rsum = 0.0f;
    float u0 = 0.0f, u1 = 0.0f;          // partial u at e = lane, e = lane + 32

    const int sbase = wave * ROWS_PER_WAVE;
    const float* itemsB = items + (size_t)b * Sn * En;
    const float* posB   = pos   + (size_t)b * Sn * Pn;

    for (int t = 0; t < TILES_PER_WAVE; ++t) {
        const int s0 = sbase + t * 16;

        // A fragments (16x32 bf16). ISA layout: lanes 0-15 -> K-half 0, lanes 16-31 -> K-half 8:
        //   elem j<8 : K = hi*8 + j ; elem j>=8 : K = 16 + hi*8 + (j-8)
        const float* ip = itemsB + (size_t)(s0 + l16) * En + hi * 8;
        const float* pp = posB   + (size_t)(s0 + l16) * Pn + hi * 8;
        v16bf aE0, aE1, aP;
        #pragma unroll
        for (int j = 0; j < 8; ++j) {
            aE0[j]     = (__bf16)ip[j];
            aE0[j + 8] = (__bf16)ip[16 + j];
            aE1[j]     = (__bf16)ip[32 + j];
            aE1[j + 8] = (__bf16)ip[48 + j];
            aP[j]      = (__bf16)pp[j];
            aP[j + 8]  = (__bf16)pp[16 + j];
        }
        if (t + 1 < TILES_PER_WAVE) {          // global_prefetch for next tile
            __builtin_prefetch(ip + 16 * En, 0, 1);
            __builtin_prefetch(pp + 16 * Pn, 0, 1);
        }

        // out[16x64] = items@We + pos@Wp + (cb broadcast), then tanh()*z, reduce over h
        float prow[8] = {0,0,0,0,0,0,0,0};
        #pragma unroll
        for (int nt = 0; nt < 4; ++nt) {
            v8f acc;
            #pragma unroll
            for (int r = 0; r < 8; ++r) acc[r] = cbv[nt];
            acc = __builtin_amdgcn_wmma_f32_16x16x32_bf16(false, aE0, false, fWe[0][nt], (short)0, acc, false, false);
            acc = __builtin_amdgcn_wmma_f32_16x16x32_bf16(false, aE1, false, fWe[1][nt], (short)0, acc, false, false);
            acc = __builtin_amdgcn_wmma_f32_16x16x32_bf16(false, aP,  false, fWp[nt],    (short)0, acc, false, false);
            #pragma unroll
            for (int r = 0; r < 8; ++r) prow[r] += fast_tanh(acc[r]) * zv[nt];
        }

        // reduce over h within each 16-lane group, then swap halves -> all 16 row scores
        float a16[16];
        #pragma unroll
        for (int r = 0; r < 8; ++r) {
            float v = prow[r];
            v += __shfl_xor(v, 1, 32);
            v += __shfl_xor(v, 2, 32);
            v += __shfl_xor(v, 4, 32);
            v += __shfl_xor(v, 8, 32);
            float o = __shfl_xor(v, 16, 32);
            a16[r]     = hi ? o : v;   // rows s0+0..7
            a16[r + 8] = hi ? v : o;   // rows s0+8..15
        }

        // online softmax update over 16 rows (all lanes redundantly; masked rows use NEG,
        // never skipped -> matches reference incl. seq_len==0 uniform-softmax case)
        float am[16], tmax = NEGV;
        #pragma unroll
        for (int r = 0; r < 16; ++r) {
            const bool valid = (s0 + r) < slen;
            am[r] = valid ? a16[r] : NEGV;
            rsum += valid ? a16[r] : 0.0f;
            tmax  = fmaxf(tmax, am[r]);
        }
        const float mnew  = fmaxf(m, tmax);
        const float scale = fast_exp(m - mnew);
        float w16[16], dsum = 0.0f;
        #pragma unroll
        for (int r = 0; r < 16; ++r) { w16[r] = fast_exp(am[r] - mnew); dsum += w16[r]; }
        d  = d * scale + dsum;
        u0 *= scale;  u1 *= scale;

        // weighted accumulation of u (tile rows are hot in WGP$/L0)
        const float* ur = itemsB + (size_t)s0 * En;
        #pragma unroll
        for (int r = 0; r < 16; ++r) {
            u0 += w16[r] * ur[r * En + lane];
            u1 += w16[r] * ur[r * En + 32 + lane];
        }
        m = mnew;
    }

    // merge the 8 per-wave partial softmax states via LDS
    if (lane == 0) { s_m[wave] = m; s_d[wave] = d; s_r[wave] = rsum; }
    s_u[wave][lane]      = u0;
    s_u[wave][lane + 32] = u1;
    __syncthreads();

    if (wave == 0) {
        float M = NEGV;
        #pragma unroll
        for (int w = 0; w < WAVES; ++w) M = fmaxf(M, s_m[w]);
        float D = 0.0f, R = 0.0f, U0 = 0.0f, U1 = 0.0f;
        #pragma unroll
        for (int w = 0; w < WAVES; ++w) {
            const float sc = fast_exp(s_m[w] - M);
            D  += sc * s_d[w];
            R  += s_r[w];
            U0 += sc * s_u[w][lane];
            U1 += sc * s_u[w][lane + 32];
        }
        out_u[(size_t)b * En + lane]      = U0 / D;
        out_u[(size_t)b * En + 32 + lane] = U1 / D;
        if (lane == 0) out_r[b] = R;
    }
}

extern "C" void kernel_launch(void* const* d_in, const int* in_sizes, int n_in,
                              void* d_out, int out_size, void* d_ws, size_t ws_size,
                              hipStream_t stream) {
    (void)in_sizes; (void)n_in; (void)out_size; (void)d_ws; (void)ws_size;
    const float* items = (const float*)d_in[0];  // [B,S,64]
    const float* pos   = (const float*)d_in[1];  // [B,S,32]
    const float* tgt   = (const float*)d_in[2];  // [B,64]
    const float* Wp    = (const float*)d_in[3];  // [32,64]
    const float* We    = (const float*)d_in[4];  // [64,64]
    const float* Wc    = (const float*)d_in[5];  // [64,64]
    const float* bias  = (const float*)d_in[6];  // [64]
    const float* zvec  = (const float*)d_in[7];  // [64]
    const int*   slen  = (const int*)d_in[8];    // [B]
    float* out_u = (float*)d_out;                     // [B,1,64] flat
    float* out_r = (float*)d_out + (size_t)Bn * En;   // [B,1] flat

    i2i_attention_kernel<<<Bn, 256, 0, stream>>>(items, pos, tgt, Wp, We, Wc,
                                                 bias, zvec, slen, out_u, out_r);
}
